// FlowerOfLifeGAT_42245298323794
// MI455X (gfx1250) — compile-verified, matrix operand order checked by reading, not run on hardware
//
#include <hip/hip_runtime.h>
#include <hip/hip_bf16.h>
#include <stdint.h>

typedef __attribute__((ext_vector_type(4)))  float  f4;
typedef __attribute__((ext_vector_type(8)))  float  v8f;
typedef __attribute__((ext_vector_type(16))) __bf16 v16bf;
typedef __attribute__((ext_vector_type(8)))  __bf16 v8bf;

#define BB     8
#define TT     4095
#define DD     1024
#define NN     7
#define GG     585          // TT / NN
#define GSPLIT 13
#define GCH    45           // GG / GSPLIT
#define MROWS  56           // BB * NN
#define MPAD   64           // padded to 4 WMMA M-tiles

// ---------------------------------------------------------------------------
// Pass 1a: partial group-mean reduction over G (streams all of x once).
// One block per (g-chunk, bn); 256 threads x float4 = one 1024-wide D row.
// ---------------------------------------------------------------------------
__global__ __launch_bounds__(256)
void folg_reduce_partial(const float* __restrict__ x, float* __restrict__ partials) {
    const int blk   = blockIdx.x;
    const int chunk = blk / MROWS;
    const int bn    = blk % MROWS;
    const int b     = bn / NN;
    const int n     = bn % NN;
    const int tid   = threadIdx.x;

    const f4* __restrict__ x4 = (const f4*)x;
    f4 acc = (f4)0.0f;
    const int g0 = chunk * GCH;
    #pragma unroll 5
    for (int g = g0; g < g0 + GCH; ++g) {
        const int row = b * TT + g * NN + n;          // (b, t) row index
        acc += x4[row * (DD / 4) + tid];
    }
    ((f4*)partials)[(chunk * MROWS + bn) * (DD / 4) + tid] = acc;
}

// ---------------------------------------------------------------------------
// Pass 1b: finalize mean, write f32 nodes (padded to 64 rows) + bf16 copy.
// ---------------------------------------------------------------------------
__global__ __launch_bounds__(256)
void folg_reduce_final(const float* __restrict__ partials,
                       float* __restrict__ nodes, __bf16* __restrict__ nodes_bf) {
    const int bn  = blockIdx.x;      // 0..63 (56..63 are zero pad rows)
    const int tid = threadIdx.x;
    f4 s = (f4)0.0f;
    if (bn < MROWS) {
        const f4* p4 = (const f4*)partials;
        #pragma unroll
        for (int c = 0; c < GSPLIT; ++c)
            s += p4[(c * MROWS + bn) * (DD / 4) + tid];
        s *= (1.0f / (float)GG);
    }
    ((f4*)nodes)[bn * (DD / 4) + tid] = s;
    const int base = bn * DD + tid * 4;
    nodes_bf[base + 0] = (__bf16)s.x;
    nodes_bf[base + 1] = (__bf16)s.y;
    nodes_bf[base + 2] = (__bf16)s.z;
    nodes_bf[base + 3] = (__bf16)s.w;
}

// ---------------------------------------------------------------------------
// f32 -> bf16 weight conversion (1M elements).
// ---------------------------------------------------------------------------
__global__ __launch_bounds__(256)
void folg_cvt_bf16(const float* __restrict__ src, __bf16* __restrict__ dst) {
    const int i = (blockIdx.x * 256 + threadIdx.x) * 4;
    f4 v = ((const f4*)src)[i >> 2];
    dst[i + 0] = (__bf16)v.x;
    dst[i + 1] = (__bf16)v.y;
    dst[i + 2] = (__bf16)v.z;
    dst[i + 3] = (__bf16)v.w;
}

// ---------------------------------------------------------------------------
// WMMA GEMM (NT):  C[m,n] = sum_k A[m,k] * Bw[n,k]
//   A : [64 x 1024] bf16 row-major, Bw : [1024 x 1024] bf16 row-major.
// One wave per 16x16 C tile; K-per-lane operand layout means both A and Bw
// load as contiguous 16B runs (no transpose needed).
// grid = (8, 4), block = 256 (8 waves -> 8 consecutive n-tiles).
// ---------------------------------------------------------------------------
__global__ __launch_bounds__(256)
void folg_wmma_gemm_nt(const __bf16* __restrict__ A, const __bf16* __restrict__ Bw,
                       float* __restrict__ C) {
    const int lane  = threadIdx.x & 31;
    const int wave  = threadIdx.x >> 5;
    const int mtile = blockIdx.y;                 // 0..3
    const int ntile = blockIdx.x * 8 + wave;      // 0..63
    const int khalf = lane >> 4;                  // 0: K[0..7],K[16..23]  1: K[8..15],K[24..31]
    const int rowA  = mtile * 16 + (lane & 15);
    const int rowB  = ntile * 16 + (lane & 15);

    const __bf16* ap = A  + rowA * DD + khalf * 8;
    const __bf16* bp = Bw + rowB * DD + khalf * 8;

    v8f c = {};
    #pragma unroll 4
    for (int k0 = 0; k0 < DD; k0 += 32) {
        v16bf a, b;
        ((v8bf*)&a)[0] = *(const v8bf*)(ap + k0);
        ((v8bf*)&a)[1] = *(const v8bf*)(ap + k0 + 16);
        ((v8bf*)&b)[0] = *(const v8bf*)(bp + k0);
        ((v8bf*)&b)[1] = *(const v8bf*)(bp + k0 + 16);
        c = __builtin_amdgcn_wmma_f32_16x16x32_bf16(false, a, false, b,
                                                    (short)0, c, false, false);
    }

    // C/D layout: VGPR r -> M = r + 8*khalf, N = lane&15
    const int col   = ntile * 16 + (lane & 15);
    const int rbase = mtile * 16 + khalf * 8;
    #pragma unroll
    for (int r = 0; r < 8; ++r)
        C[(rbase + r) * DD + col] = c[r];
}

// ---------------------------------------------------------------------------
// Attention scores: s1[bn] = h[bn,:]·a1,  s2[bn] = h[bn,:]·a2
// ---------------------------------------------------------------------------
__global__ __launch_bounds__(256)
void folg_scores(const float* __restrict__ h, const float* __restrict__ a_w,
                 float* __restrict__ s12) {
    __shared__ float sh1[256], sh2[256];
    const int bn  = blockIdx.x;
    const int tid = threadIdx.x;
    float p1 = 0.f, p2 = 0.f;
    #pragma unroll 4
    for (int d = tid; d < DD; d += 256) {
        const float hv = h[bn * DD + d];
        p1 += hv * a_w[d];
        p2 += hv * a_w[DD + d];
    }
    sh1[tid] = p1; sh2[tid] = p2;
    __syncthreads();
    for (int s = 128; s > 0; s >>= 1) {
        if (tid < s) { sh1[tid] += sh1[tid + s]; sh2[tid] += sh2[tid + s]; }
        __syncthreads();
    }
    if (tid == 0) { s12[bn] = sh1[0]; s12[MROWS + bn] = sh2[0]; }
}

// Flower-of-life adjacency: node 0 <-> 1..6; ring 1-2-3-4-5-6-1.
__device__ __forceinline__ bool folg_adj(int i, int j) {
    if (i == j) return false;
    if (i == 0 || j == 0) return true;
    return (j == (i % 6) + 1) || (i == (j % 6) + 1);
}

// ---------------------------------------------------------------------------
// Masked softmax over 7x7 per batch. grid = 8, block = 64 (7 rows active).
// ---------------------------------------------------------------------------
__global__ __launch_bounds__(64)
void folg_softmax(const float* __restrict__ s12, float* __restrict__ alpha) {
    const int b = blockIdx.x;
    const int i = threadIdx.x;
    if (i >= NN) return;
    float e[NN];
    float mx = -3.0e38f;
    #pragma unroll
    for (int j = 0; j < NN; ++j) {
        e[j] = folg_adj(i, j) ? (s12[b * NN + i] + s12[MROWS + b * NN + j]) : -3.0e38f;
        mx = fmaxf(mx, e[j]);
    }
    float sum = 0.f;
    #pragma unroll
    for (int j = 0; j < NN; ++j) {
        e[j] = folg_adj(i, j) ? __expf(e[j] - mx) : 0.0f;
        sum += e[j];
    }
    const float inv = 1.0f / sum;
    #pragma unroll
    for (int j = 0; j < NN; ++j)
        alpha[b * (NN * NN) + i * NN + j] = e[j] * inv;
}

// ---------------------------------------------------------------------------
// t = alpha @ h  (7 MACs per element) + bf16 copy, padded to 64 rows.
// ---------------------------------------------------------------------------
__global__ __launch_bounds__(256)
void folg_alphah(const float* __restrict__ h, const float* __restrict__ alpha,
                 float* __restrict__ t, __bf16* __restrict__ t_bf) {
    const int bn  = blockIdx.x;      // 0..63
    const int tid = threadIdx.x;
    f4 acc = (f4)0.0f;
    if (bn < MROWS) {
        const int b = bn / NN, n = bn % NN;
        const f4* h4 = (const f4*)h;
        #pragma unroll
        for (int j = 0; j < NN; ++j) {
            const float al = alpha[b * (NN * NN) + n * NN + j];
            acc += al * h4[(b * NN + j) * (DD / 4) + tid];
        }
    }
    ((f4*)t)[bn * (DD / 4) + tid] = acc;
    const int base = bn * DD + tid * 4;
    t_bf[base + 0] = (__bf16)acc.x;
    t_bf[base + 1] = (__bf16)acc.y;
    t_bf[base + 2] = (__bf16)acc.z;
    t_bf[base + 3] = (__bf16)acc.w;
}

// ---------------------------------------------------------------------------
// Final broadcast add: out[b,t,d] = x[b,t,d] + m[b*7 + t%7, d]
// x re-read should hit the 192MB L2; output stored non-temporally so the
// 134MB write stream doesn't evict x.
// grid = 8*4095 blocks (one (b,t) row each), block = 256 x float4.
// ---------------------------------------------------------------------------
__global__ __launch_bounds__(256)
void folg_final_add(const float* __restrict__ x, const float* __restrict__ m,
                    float* __restrict__ out) {
    const int row = blockIdx.x;          // 0..BB*TT-1
    const int b   = row / TT;
    const int t   = row % TT;
    const int n   = t % NN;
    const int tid = threadIdx.x;

    const f4 xv = ((const f4*)x)[row * (DD / 4) + tid];
    const f4 mv = ((const f4*)m)[(b * NN + n) * (DD / 4) + tid];
    const f4 o  = xv + mv;
    __builtin_nontemporal_store(o, ((f4*)out) + row * (DD / 4) + tid);
}

// ---------------------------------------------------------------------------
extern "C" void kernel_launch(void* const* d_in, const int* in_sizes, int n_in,
                              void* d_out, int out_size, void* d_ws, size_t ws_size,
                              hipStream_t stream) {
    const float* x     = (const float*)d_in[0];   // (8, 4095, 1024)
    const float* W     = (const float*)d_in[1];   // (1024, 1024)
    const float* a_w   = (const float*)d_in[2];   // (2048,)
    const float* out_W = (const float*)d_in[3];   // (1024, 1024)
    float*       out   = (float*)d_out;

    // ---- workspace carve-up (f32 region, then 256B-aligned bf16 region) ----
    float* ws       = (float*)d_ws;
    float* partials = ws;                                   // 13*56*1024
    float* nodes    = partials + GSPLIT * MROWS * DD;       // 64*1024
    float* h        = nodes + MPAD * DD;                    // 64*1024
    float* t        = h + MPAD * DD;                        // 64*1024
    float* m        = t + MPAD * DD;                        // 64*1024
    float* s12      = m + MPAD * DD;                        // 112
    float* alpha    = s12 + 2 * MROWS;                      // 392

    uintptr_t pb = (uintptr_t)(alpha + BB * NN * NN);
    pb = (pb + 255) & ~(uintptr_t)255;
    __bf16* nodes_bf = (__bf16*)pb;                         // 64*1024
    __bf16* W_bf     = nodes_bf + MPAD * DD;                // 1024*1024
    __bf16* oW_bf    = W_bf + DD * DD;                      // 1024*1024
    __bf16* t_bf     = oW_bf + DD * DD;                     // 64*1024

    // 1. group-mean reduction (streams x: 134 MB -> lands in L2)
    folg_reduce_partial<<<GSPLIT * MROWS, 256, 0, stream>>>(x, partials);
    folg_reduce_final<<<MPAD, 256, 0, stream>>>(partials, nodes, nodes_bf);

    // 2. weight conversion to bf16
    folg_cvt_bf16<<<(DD * DD) / 1024, 256, 0, stream>>>(W, W_bf);
    folg_cvt_bf16<<<(DD * DD) / 1024, 256, 0, stream>>>(out_W, oW_bf);

    // 3. h = nodes @ W^T   (WMMA bf16, f32 accumulate)
    folg_wmma_gemm_nt<<<dim3(8, 4), 256, 0, stream>>>(nodes_bf, W_bf, h);

    // 4. attention scores + masked softmax
    folg_scores<<<MROWS, 256, 0, stream>>>(h, a_w, s12);
    folg_softmax<<<BB, 64, 0, stream>>>(s12, alpha);

    // 5. t = alpha @ h, then m = t @ out_W^T (WMMA)
    folg_alphah<<<MPAD, 256, 0, stream>>>(h, alpha, t, t_bf);
    folg_wmma_gemm_nt<<<dim3(8, 4), 256, 0, stream>>>(t_bf, oW_bf, m);

    // 6. out = x + broadcast(m)  (x from L2, NT store for output)
    folg_final_add<<<BB * TT, 256, 0, stream>>>(x, m, out);
}
